// PointNet2Multiview2_7954279432400
// MI455X (gfx1250) — compile-verified, matrix operand order checked by reading, not run on hardware
//
#include <hip/hip_runtime.h>
#include <stdint.h>

// ---------------------------------------------------------------------------
// PointNet++ multiview forward for MI455X (gfx1250, wave32).
// All channel-mixing layers run through v_wmma_f32_16x16x32_f16.
// ---------------------------------------------------------------------------

typedef __attribute__((ext_vector_type(16))) _Float16 v16h;
typedef __attribute__((ext_vector_type(8)))  _Float16 v8h;
typedef __attribute__((ext_vector_type(8)))  float    v8f;

static constexpr int BB   = 4;
static constexpr int NPTS = 8192;
static constexpr int VV   = 3;
static constexpr int FDIM = 128;
static constexpr int NCLS = 21;

// ---------------------------------------------------------------------------
// xyz (B,3,N) -> xyzT (B,N,3)
// ---------------------------------------------------------------------------
__global__ void k_xyzT(const float* __restrict__ xyz, float* __restrict__ out, int total) {
  int g = blockIdx.x * blockDim.x + threadIdx.x;
  if (g >= total) return;
  int b = g / NPTS, n = g % NPTS;
  const float* src = xyz + (size_t)b * 3 * NPTS + n;
  float* o = out + (size_t)g * 3;
  o[0] = src[0]; o[1] = src[(size_t)NPTS]; o[2] = src[(size_t)2 * NPTS];
}

// ---------------------------------------------------------------------------
// Multiview select: first view whose per-point channel column is not all-zero.
// image_feats (B,V,F,N) f32 -> out (B,N,F) f16
// ---------------------------------------------------------------------------
__global__ void k_select(const float* __restrict__ img, _Float16* __restrict__ out, int total) {
  int g = blockIdx.x * blockDim.x + threadIdx.x;
  if (g >= total) return;
  int b = g / NPTS, n = g % NPTS;
  const float* base = img + (size_t)b * VV * FDIM * NPTS;
  int pick = VV - 1;
  for (int v = 0; v < VV; ++v) {
    const float* vb = base + (size_t)v * FDIM * NPTS + n;
    bool any = false;
    for (int c = 0; c < FDIM; ++c)
      if (vb[(size_t)c * NPTS] != 0.f) { any = true; break; }
    if (any) { pick = v; break; }
  }
  const float* vb = base + (size_t)pick * FDIM * NPTS + n;
  _Float16* o = out + (size_t)g * FDIM;
  for (int c = 0; c < FDIM; ++c) o[c] = (_Float16)vb[(size_t)c * NPTS];
}

// ---------------------------------------------------------------------------
// Farthest point sampling. One 256-thread block per batch. Coords + running
// min-distance live in registers; argmax via LDS tree (tie -> lowest index,
// matching jnp.argmax). Output idx sequence matches lax.scan (first idx = 0).
// ---------------------------------------------------------------------------
template <int PPT>
__global__ void __launch_bounds__(256) k_fps(const float* __restrict__ pts, int n, int npoint,
                                             int* __restrict__ out) {
  __shared__ float sv[256];
  __shared__ int   si[256];
  int b = blockIdx.x;
  int t = threadIdx.x;
  const float* P = pts + (size_t)b * n * 3;
  float px[PPT], py[PPT], pz[PPT], dist[PPT];
#pragma unroll
  for (int i = 0; i < PPT; ++i) {
    int gi = t + i * 256;
    dist[i] = 1e10f;
    if (gi < n) { px[i] = P[gi * 3]; py[i] = P[gi * 3 + 1]; pz[i] = P[gi * 3 + 2]; }
    else        { px[i] = 0.f; py[i] = 0.f; pz[i] = 0.f; }
  }
  int far = 0;
  for (int s = 0; s < npoint; ++s) {
    if (t == 0) out[b * npoint + s] = far;
    float fx = P[far * 3], fy = P[far * 3 + 1], fz = P[far * 3 + 2];
    float best = -1.f; int bi = 0x7fffffff;
#pragma unroll
    for (int i = 0; i < PPT; ++i) {
      int gi = t + i * 256;
      if (gi < n) {
        float dx = px[i] - fx, dy = py[i] - fy, dz = pz[i] - fz;
        float d = dx * dx + dy * dy + dz * dz;
        float nd = fminf(dist[i], d);
        dist[i] = nd;
        if (nd > best || (nd == best && gi < bi)) { best = nd; bi = gi; }
      }
    }
    sv[t] = best; si[t] = bi;
    __syncthreads();
    for (int o = 128; o > 0; o >>= 1) {
      if (t < o) {
        if (sv[t + o] > sv[t] || (sv[t + o] == sv[t] && si[t + o] < si[t])) {
          sv[t] = sv[t + o]; si[t] = si[t + o];
        }
      }
      __syncthreads();
    }
    far = si[0];
    __syncthreads();
  }
}

// ---------------------------------------------------------------------------
// new_xyz gather
// ---------------------------------------------------------------------------
__global__ void k_gatherxyz(const float* __restrict__ pts, const int* __restrict__ idx,
                            float* __restrict__ out, int n, int m, int total) {
  int g = blockIdx.x * blockDim.x + threadIdx.x;
  if (g >= total) return;
  int b = g / m;
  int pi = idx[g];
  const float* p = pts + ((size_t)b * n + pi) * 3;
  float* o = out + (size_t)g * 3;
  o[0] = p[0]; o[1] = p[1]; o[2] = p[2];
}

// ---------------------------------------------------------------------------
// Ball query: first 32 indices (ascending) within radius; remainder = first.
// ---------------------------------------------------------------------------
__global__ void k_ballq(const float* __restrict__ pts, const float* __restrict__ nx,
                        int n, int m, float r2, int* __restrict__ out, int total) {
  int g = blockIdx.x * blockDim.x + threadIdx.x;
  if (g >= total) return;
  int b = g / m;
  const float* P = pts + (size_t)b * n * 3;
  const float* q = nx + (size_t)g * 3;
  float qx = q[0], qy = q[1], qz = q[2];
  int* o = out + (size_t)g * 32;
  int cnt = 0;
  for (int i = 0; i < n && cnt < 32; ++i) {
    float dx = P[i * 3] - qx, dy = P[i * 3 + 1] - qy, dz = P[i * 3 + 2] - qz;
    if (dx * dx + dy * dy + dz * dz <= r2) o[cnt++] = i;
  }
  int first = (cnt > 0) ? o[0] : (n - 1);
  for (int i = cnt; i < 32; ++i) o[i] = first;
}

// ---------------------------------------------------------------------------
// Build grouped rows: [relative xyz (3) | feats (C)] zero-padded to Kpad (f16).
// One wave per row, lanes stride channels.
// ---------------------------------------------------------------------------
__global__ void __launch_bounds__(256) k_group(const float* __restrict__ pts,
                                               const float* __restrict__ nx,
                                               const int* __restrict__ bidx,
                                               const _Float16* __restrict__ feats,
                                               int n, int m, int C, int Kpad,
                                               _Float16* __restrict__ out, int nrows) {
  int row = blockIdx.x * 8 + (threadIdx.x >> 5);
  if (row >= nrows) return;
  int lane = threadIdx.x & 31;
  int b = row / (m * 32);
  int rem = row % (m * 32);
  int q = rem >> 5;
  int pi = bidx[row];
  const float* pp = pts + ((size_t)b * n + pi) * 3;
  const float* qq = nx + ((size_t)b * m + q) * 3;
  _Float16* o = out + (size_t)row * Kpad;
  if (lane < 3) o[lane] = (_Float16)(pp[lane] - qq[lane]);
  if (feats) {
    const _Float16* f = feats + ((size_t)b * n + pi) * C;
    for (int c = lane; c < C; c += 32) o[3 + c] = f[c];
  }
  for (int k = 3 + C + lane; k < Kpad; k += 32) o[k] = (_Float16)0.f;
}

// ---------------------------------------------------------------------------
// 3-NN search (brute force) + normalized inverse-distance weights.
// ---------------------------------------------------------------------------
__global__ void k_3nn(const float* __restrict__ x1, const float* __restrict__ x2,
                      int m1, int m2, int* __restrict__ oi, float* __restrict__ ow, int total) {
  int g = blockIdx.x * blockDim.x + threadIdx.x;
  if (g >= total) return;
  int b = g / m1;
  const float* q = x1 + (size_t)g * 3;
  const float* P = x2 + (size_t)b * m2 * 3;
  float qx = q[0], qy = q[1], qz = q[2];
  float d0 = 3e38f, d1 = 3e38f, d2 = 3e38f;
  int i0 = 0, i1 = 0, i2 = 0;
  for (int j = 0; j < m2; ++j) {
    float dx = P[j * 3] - qx, dy = P[j * 3 + 1] - qy, dz = P[j * 3 + 2] - qz;
    float d = dx * dx + dy * dy + dz * dz;
    if (d < d0)      { d2 = d1; i2 = i1; d1 = d0; i1 = i0; d0 = d; i0 = j; }
    else if (d < d1) { d2 = d1; i2 = i1; d1 = d;  i1 = j; }
    else if (d < d2) { d2 = d;  i2 = j; }
  }
  float w0 = 1.f / (d0 + 1e-8f), w1 = 1.f / (d1 + 1e-8f), w2 = 1.f / (d2 + 1e-8f);
  float wsum = w0 + w1 + w2;
  oi[g * 3] = i0; oi[g * 3 + 1] = i1; oi[g * 3 + 2] = i2;
  ow[g * 3] = w0 / wsum; ow[g * 3 + 1] = w1 / wsum; ow[g * 3 + 2] = w2 / wsum;
}

// ---------------------------------------------------------------------------
// FP gather: row = [points1 (C1) | 3-NN interp of points2 (C2)] pad to Kpad.
// One wave per row.
// ---------------------------------------------------------------------------
__global__ void __launch_bounds__(256) k_fpgather(const _Float16* __restrict__ p1, int C1,
                                                  const _Float16* __restrict__ p2, int C2,
                                                  const int* __restrict__ oi,
                                                  const float* __restrict__ ow,
                                                  int m1, int m2, int Kpad,
                                                  _Float16* __restrict__ out, int nrows) {
  int row = blockIdx.x * 8 + (threadIdx.x >> 5);
  if (row >= nrows) return;
  int lane = threadIdx.x & 31;
  int b = row / m1, q = row % m1;
  int i0 = oi[row * 3], i1 = oi[row * 3 + 1], i2 = oi[row * 3 + 2];
  float w0 = ow[row * 3], w1 = ow[row * 3 + 1], w2 = ow[row * 3 + 2];
  _Float16* o = out + (size_t)row * Kpad;
  if (p1) {
    const _Float16* f = p1 + ((size_t)b * m1 + q) * C1;
    for (int c = lane; c < C1; c += 32) o[c] = f[c];
  }
  const _Float16* f0 = p2 + ((size_t)b * m2 + i0) * C2;
  const _Float16* f1 = p2 + ((size_t)b * m2 + i1) * C2;
  const _Float16* f2 = p2 + ((size_t)b * m2 + i2) * C2;
  for (int c = lane; c < C2; c += 32)
    o[C1 + c] = (_Float16)(w0 * (float)f0[c] + w1 * (float)f1[c] + w2 * (float)f2[c]);
  for (int k = C1 + C2 + lane; k < Kpad; k += 32) o[k] = (_Float16)0.f;
}

// ---------------------------------------------------------------------------
// Max over 32 samples per query point.
// ---------------------------------------------------------------------------
__global__ void k_maxpool(const _Float16* __restrict__ Y, int ldy, int C,
                          _Float16* __restrict__ out, int ldo, int co, int total) {
  int g = blockIdx.x * blockDim.x + threadIdx.x;
  if (g >= total) return;
  int bm = g / C, c = g % C;
  const _Float16* p = Y + (size_t)bm * 32 * ldy + c;
  float mx = -3e38f;
#pragma unroll 8
  for (int s = 0; s < 32; ++s) mx = fmaxf(mx, (float)p[(size_t)s * ldy]);
  out[(size_t)bm * ldo + co + c] = (_Float16)mx;
}

// ---------------------------------------------------------------------------
// Weight prep: W f32 [Kreal][CoutReal] -> Wt f16 [CoutPad][Kpad] (= B^T, padded)
// ---------------------------------------------------------------------------
__global__ void k_prepw(const float* __restrict__ W, _Float16* __restrict__ Wt,
                        int Kreal, int CoutReal, int Kpad, int CoutPad, int total) {
  int g = blockIdx.x * blockDim.x + threadIdx.x;
  if (g >= total) return;
  int co = g / Kpad, k = g % Kpad;
  float v = (co < CoutReal && k < Kreal) ? W[(size_t)k * CoutReal + co] : 0.f;
  Wt[g] = (_Float16)v;
}

// ---------------------------------------------------------------------------
// WMMA GEMM + bias (+affine) (+relu). Block: 256 threads = 8 waves.
// Block tile 64(M) x 32(N); each wave one 16x16 f32 accumulator, K chunks of 32.
// Fragment layout per CDNA5 ISA 16-bit A-matrix table:
//   row (M for A, N for B^T) = lane&15; lane<16 holds K {0..7,16..23} (+k0),
//   lane>=16 holds K {8..15,24..31} -> two contiguous 16B loads per lane.
// C/D layout: N = lane&15, M = r + 8*(lane>=16).
// ---------------------------------------------------------------------------
template <bool RELU, bool AFFINE, bool OUTF32>
__global__ void __launch_bounds__(256) k_mlp(const _Float16* __restrict__ X, int ldx,
                                             const _Float16* __restrict__ Wt,
                                             const float* __restrict__ bias,
                                             const float* __restrict__ scale,
                                             const float* __restrict__ shift,
                                             void* __restrict__ Yv, int ldy, int ncols, int K) {
  const int lane = threadIdx.x & 31;
  const int wave = threadIdx.x >> 5;
  const int m0 = blockIdx.y * 64 + (wave & 3) * 16;
  const int n0 = blockIdx.x * 32 + (wave >> 2) * 16;
  const int lm = lane & 15;
  const int lb = (lane >> 4) << 3;
  const _Float16* xr = X  + (size_t)(m0 + lm) * ldx + lb;
  const _Float16* wr = Wt + (size_t)(n0 + lm) * ldx + lb;
  v8f acc = {0.f, 0.f, 0.f, 0.f, 0.f, 0.f, 0.f, 0.f};
  for (int k = 0; k < K; k += 32) {
    v8h a0 = *(const v8h*)(xr + k);
    v8h a1 = *(const v8h*)(xr + k + 16);
    v8h b0 = *(const v8h*)(wr + k);
    v8h b1 = *(const v8h*)(wr + k + 16);
    v16h av, bv;
#pragma unroll
    for (int i = 0; i < 8; ++i) { av[i] = a0[i]; av[i + 8] = a1[i]; bv[i] = b0[i]; bv[i + 8] = b1[i]; }
    acc = __builtin_amdgcn_wmma_f32_16x16x32_f16(false, av, false, bv, (short)0, acc, false, false);
  }
  const int col = n0 + lm;
  float bb = 0.f, sc = 1.f, sh = 0.f;
  if (col < ncols) {
    bb = bias[col];
    if (AFFINE) { sc = scale[col]; sh = shift[col]; }
  }
  const int mhi = (lane >> 4) << 3;
#pragma unroll
  for (int r = 0; r < 8; ++r) {
    if (col < ncols) {
      float v = acc[r] + bb;
      if (AFFINE) v = v * sc + sh;
      if (RELU)   v = fmaxf(v, 0.f);
      int row = m0 + mhi + r;
      if (OUTF32) ((float*)Yv)[(size_t)row * ldy + col] = v;
      else        ((_Float16*)Yv)[(size_t)row * ldy + col] = (_Float16)v;
    }
  }
}

// ---------------------------------------------------------------------------
// Host orchestration
// ---------------------------------------------------------------------------
extern "C" void kernel_launch(void* const* d_in, const int* in_sizes, int n_in,
                              void* d_out, int out_size, void* d_ws, size_t ws_size,
                              hipStream_t stream) {
  (void)in_sizes; (void)n_in; (void)out_size; (void)ws_size;
  const float* xyz = (const float*)d_in[0];
  const float* img = (const float*)d_in[1];
  auto pf = [&](int i) { return (const float*)d_in[2 + i]; };

  // param index map (offsets after the first 2 inputs)
  const int SA1G = 0, SA2G = 12, SA1F = 24, SA2F = 36, SA3 = 48, SA4 = 60;
  const int FP4 = 72, FP3 = 80, FP2 = 88, FP1 = 96;
  const int C1W = 108, C1B = 109, BN1S = 110, BN1T = 111, C2W = 112, C2B = 113;

  // ---- workspace carving (deterministic) ----
  uint8_t* wsb = (uint8_t*)d_ws;
  size_t off = 0;
  auto alloc = [&](size_t bytes) -> void* {
    void* p = wsb + off;
    off += (bytes + 255) & ~(size_t)255;
    return p;
  };

  float*     xyzT   = (float*)    alloc((size_t)BB * NPTS * 3 * 4);
  _Float16*  imgf   = (_Float16*) alloc((size_t)BB * NPTS * FDIM * 2);
  _Float16*  ping   = (_Float16*) alloc((size_t)131072 * 160 * 2);
  _Float16*  pong   = (_Float16*) alloc((size_t)131072 * 64 * 2);
  int*       fpsidx = (int*)      alloc((size_t)BB * 1024 * 4);
  int*       ballix = (int*)      alloc((size_t)BB * 1024 * 32 * 4);
  float*     l1xyz  = (float*)    alloc((size_t)BB * 1024 * 3 * 4);
  float*     l2xyz  = (float*)    alloc((size_t)BB * 256 * 3 * 4);
  float*     l3xyz  = (float*)    alloc((size_t)BB * 64 * 3 * 4);
  float*     l4xyz  = (float*)    alloc((size_t)BB * 16 * 3 * 4);
  _Float16*  l1pg   = (_Float16*) alloc((size_t)BB * 1024 * 64 * 2);
  _Float16*  l1pf   = (_Float16*) alloc((size_t)BB * 1024 * 64 * 2);
  _Float16*  l2p    = (_Float16*) alloc((size_t)BB * 256 * 256 * 2);
  _Float16*  l3p    = (_Float16*) alloc((size_t)BB * 64 * 256 * 2);
  _Float16*  l4p    = (_Float16*) alloc((size_t)BB * 16 * 512 * 2);
  _Float16*  fp4o   = (_Float16*) alloc((size_t)BB * 64 * 256 * 2);
  _Float16*  fp3o   = (_Float16*) alloc((size_t)BB * 256 * 256 * 2);
  _Float16*  fp2o   = (_Float16*) alloc((size_t)BB * 1024 * 128 * 2);
  int*       nnidx  = (int*)      alloc((size_t)BB * NPTS * 3 * 4);
  float*     nnw    = (float*)    alloc((size_t)BB * NPTS * 3 * 4);

  // GEMM helper: prep transposed f16 weights then launch WMMA MLP layer
  auto mlp = [&](const _Float16* X, int Kpad, const float* W, const float* b,
                 const float* s, const float* t, int Kreal, int CoutReal, int CoutPad,
                 void* Y, int ldy, int ncols, int rows, bool relu, bool f32out) {
    _Float16* Wt = (_Float16*)alloc((size_t)CoutPad * Kpad * 2);
    int tot = CoutPad * Kpad;
    k_prepw<<<(tot + 255) / 256, 256, 0, stream>>>(W, Wt, Kreal, CoutReal, Kpad, CoutPad, tot);
    dim3 g(CoutPad / 32, rows / 64), blk(256);
    if (f32out)
      k_mlp<false, false, true><<<g, blk, 0, stream>>>(X, Kpad, Wt, b, nullptr, nullptr, Y, ldy, ncols, Kpad);
    else if (relu)
      k_mlp<true, true, false><<<g, blk, 0, stream>>>(X, Kpad, Wt, b, s, t, Y, ldy, ncols, Kpad);
    else
      k_mlp<false, false, false><<<g, blk, 0, stream>>>(X, Kpad, Wt, b, nullptr, nullptr, Y, ldy, ncols, Kpad);
  };

  // 3-layer SA MLP chain: ping -> pong -> ping -> pong
  auto sa_mlp3 = [&](int pb, int K0real, int K0pad, int c0, int c1, int c2, int rows) {
    mlp(ping, K0pad, pf(pb + 0), pf(pb + 1), pf(pb + 2),  pf(pb + 3),  K0real, c0, c0, pong, c0, c0, rows, true, false);
    mlp(pong, c0,    pf(pb + 4), pf(pb + 5), pf(pb + 6),  pf(pb + 7),  c0,     c1, c1, ping, c1, c1, rows, true, false);
    mlp(ping, c1,    pf(pb + 8), pf(pb + 9), pf(pb + 10), pf(pb + 11), c1,     c2, c2, pong, c2, c2, rows, true, false);
  };

  const int T = 256;

  // ---- preprocessing ----
  k_xyzT<<<(BB * NPTS + T - 1) / T, T, 0, stream>>>(xyz, xyzT, BB * NPTS);
  k_select<<<(BB * NPTS + T - 1) / T, T, 0, stream>>>(img, imgf, BB * NPTS);

  // ---- Level 1 geometry (shared by geo & feat branches) ----
  k_fps<32><<<BB, 256, 0, stream>>>(xyzT, NPTS, 1024, fpsidx);
  k_gatherxyz<<<(BB * 1024 + T - 1) / T, T, 0, stream>>>(xyzT, fpsidx, l1xyz, NPTS, 1024, BB * 1024);
  k_ballq<<<(BB * 1024 + T - 1) / T, T, 0, stream>>>(xyzT, l1xyz, NPTS, 1024, 0.01f, ballix, BB * 1024);
  const int rows1 = BB * 1024 * 32;

  // sa1_geo: 3 -> 32 -> 32 -> 64
  k_group<<<rows1 / 8, 256, 0, stream>>>(xyzT, l1xyz, ballix, nullptr, NPTS, 1024, 0, 32, ping, rows1);
  sa_mlp3(SA1G, 3, 32, 32, 32, 64, rows1);
  k_maxpool<<<(BB * 1024 * 64 + T - 1) / T, T, 0, stream>>>(pong, 64, 64, l1pg, 64, 0, BB * 1024 * 64);

  // sa1_feat: 131 -> 32 -> 32 -> 64
  k_group<<<rows1 / 8, 256, 0, stream>>>(xyzT, l1xyz, ballix, imgf, NPTS, 1024, FDIM, 160, ping, rows1);
  sa_mlp3(SA1F, 131, 160, 32, 32, 64, rows1);
  k_maxpool<<<(BB * 1024 * 64 + T - 1) / T, T, 0, stream>>>(pong, 64, 64, l1pf, 64, 0, BB * 1024 * 64);

  // ---- Level 2 ----
  k_fps<4><<<BB, 256, 0, stream>>>(l1xyz, 1024, 256, fpsidx);
  k_gatherxyz<<<(BB * 256 + T - 1) / T, T, 0, stream>>>(l1xyz, fpsidx, l2xyz, 1024, 256, BB * 256);
  k_ballq<<<(BB * 256 + T - 1) / T, T, 0, stream>>>(l1xyz, l2xyz, 1024, 256, 0.04f, ballix, BB * 256);
  const int rows2 = BB * 256 * 32;

  // sa2_geo: 67 -> 64 -> 64 -> 128  (to l2p channels [0,128))
  k_group<<<rows2 / 8, 256, 0, stream>>>(l1xyz, l2xyz, ballix, l1pg, 1024, 256, 64, 96, ping, rows2);
  sa_mlp3(SA2G, 67, 96, 64, 64, 128, rows2);
  k_maxpool<<<(BB * 256 * 128 + T - 1) / T, T, 0, stream>>>(pong, 128, 128, l2p, 256, 0, BB * 256 * 128);

  // sa2_feat: 67 -> 64 -> 64 -> 128  (to l2p channels [128,256))
  k_group<<<rows2 / 8, 256, 0, stream>>>(l1xyz, l2xyz, ballix, l1pf, 1024, 256, 64, 96, ping, rows2);
  sa_mlp3(SA2F, 67, 96, 64, 64, 128, rows2);
  k_maxpool<<<(BB * 256 * 128 + T - 1) / T, T, 0, stream>>>(pong, 128, 128, l2p, 256, 128, BB * 256 * 128);

  // ---- Level 3: sa3 259 -> 128 -> 128 -> 256 ----
  k_fps<1><<<BB, 256, 0, stream>>>(l2xyz, 256, 64, fpsidx);
  k_gatherxyz<<<(BB * 64 + T - 1) / T, T, 0, stream>>>(l2xyz, fpsidx, l3xyz, 256, 64, BB * 64);
  k_ballq<<<(BB * 64 + T - 1) / T, T, 0, stream>>>(l2xyz, l3xyz, 256, 64, 0.16f, ballix, BB * 64);
  const int rows3 = BB * 64 * 32;
  k_group<<<rows3 / 8, 256, 0, stream>>>(l2xyz, l3xyz, ballix, l2p, 256, 64, 256, 288, ping, rows3);
  sa_mlp3(SA3, 259, 288, 128, 128, 256, rows3);
  k_maxpool<<<(BB * 64 * 256 + T - 1) / T, T, 0, stream>>>(pong, 256, 256, l3p, 256, 0, BB * 64 * 256);

  // ---- Level 4: sa4 259 -> 256 -> 256 -> 512 ----
  k_fps<1><<<BB, 256, 0, stream>>>(l3xyz, 64, 16, fpsidx);
  k_gatherxyz<<<(BB * 16 + T - 1) / T, T, 0, stream>>>(l3xyz, fpsidx, l4xyz, 64, 16, BB * 16);
  k_ballq<<<(BB * 16 + T - 1) / T, T, 0, stream>>>(l3xyz, l4xyz, 64, 16, 0.64f, ballix, BB * 16);
  const int rows4 = BB * 16 * 32;
  k_group<<<rows4 / 8, 256, 0, stream>>>(l3xyz, l4xyz, ballix, l3p, 64, 16, 256, 288, ping, rows4);
  sa_mlp3(SA4, 259, 288, 256, 256, 512, rows4);
  k_maxpool<<<(BB * 16 * 512 + T - 1) / T, T, 0, stream>>>(pong, 512, 512, l4p, 512, 0, BB * 16 * 512);

  // ---- FP4: (l3, l4) K=768 -> 256 -> 256 ----
  k_3nn<<<(BB * 64 + T - 1) / T, T, 0, stream>>>(l3xyz, l4xyz, 64, 16, nnidx, nnw, BB * 64);
  k_fpgather<<<(BB * 64) / 8, 256, 0, stream>>>(l3p, 256, l4p, 512, nnidx, nnw, 64, 16, 768, ping, BB * 64);
  mlp(ping, 768, pf(FP4 + 0), pf(FP4 + 1), pf(FP4 + 2), pf(FP4 + 3), 768, 256, 256, pong, 256, 256, BB * 64, true, false);
  mlp(pong, 256, pf(FP4 + 4), pf(FP4 + 5), pf(FP4 + 6), pf(FP4 + 7), 256, 256, 256, fp4o, 256, 256, BB * 64, true, false);

  // ---- FP3: (l2, l3) K=512 -> 256 -> 256 ----
  k_3nn<<<(BB * 256 + T - 1) / T, T, 0, stream>>>(l2xyz, l3xyz, 256, 64, nnidx, nnw, BB * 256);
  k_fpgather<<<(BB * 256) / 8, 256, 0, stream>>>(l2p, 256, fp4o, 256, nnidx, nnw, 256, 64, 512, ping, BB * 256);
  mlp(ping, 512, pf(FP3 + 0), pf(FP3 + 1), pf(FP3 + 2), pf(FP3 + 3), 512, 256, 256, pong, 256, 256, BB * 256, true, false);
  mlp(pong, 256, pf(FP3 + 4), pf(FP3 + 5), pf(FP3 + 6), pf(FP3 + 7), 256, 256, 256, fp3o, 256, 256, BB * 256, true, false);

  // ---- FP2: (l1, l2) K=320 -> 256 -> 128 ----
  k_3nn<<<(BB * 1024 + T - 1) / T, T, 0, stream>>>(l1xyz, l2xyz, 1024, 256, nnidx, nnw, BB * 1024);
  k_fpgather<<<(BB * 1024) / 8, 256, 0, stream>>>(l1pg, 64, fp3o, 256, nnidx, nnw, 1024, 256, 320, ping, BB * 1024);
  mlp(ping, 320, pf(FP2 + 0), pf(FP2 + 1), pf(FP2 + 2), pf(FP2 + 3), 320, 256, 256, pong, 256, 256, BB * 1024, true, false);
  mlp(pong, 256, pf(FP2 + 4), pf(FP2 + 5), pf(FP2 + 6), pf(FP2 + 7), 256, 128, 128, fp2o, 128, 128, BB * 1024, true, false);

  // ---- FP1: (l0, l1) K=128 -> 128 -> 128 -> 128 ----
  const int rows0 = BB * NPTS;
  k_3nn<<<(rows0 + T - 1) / T, T, 0, stream>>>(xyzT, l1xyz, NPTS, 1024, nnidx, nnw, rows0);
  k_fpgather<<<rows0 / 8, 256, 0, stream>>>(nullptr, 0, fp2o, 128, nnidx, nnw, NPTS, 1024, 128, ping, rows0);
  mlp(ping, 128, pf(FP1 + 0), pf(FP1 + 1), pf(FP1 + 2),  pf(FP1 + 3),  128, 128, 128, pong, 128, 128, rows0, true, false);
  mlp(pong, 128, pf(FP1 + 4), pf(FP1 + 5), pf(FP1 + 6),  pf(FP1 + 7),  128, 128, 128, ping, 128, 128, rows0, true, false);
  mlp(ping, 128, pf(FP1 + 8), pf(FP1 + 9), pf(FP1 + 10), pf(FP1 + 11), 128, 128, 128, pong, 128, 128, rows0, true, false);

  // ---- Head: conv1 + bn1 + relu, then conv2 (f32 out, 21 classes) ----
  mlp(pong, 128, pf(C1W), pf(C1B), pf(BN1S), pf(BN1T), 128, 128, 128, ping, 128, 128, rows0, true, false);
  mlp(ping, 128, pf(C2W), pf(C2B), nullptr, nullptr, 128, NCLS, 32, d_out, NCLS, NCLS, rows0, false, true);
}
